// SEAC_32272384262899
// MI455X (gfx1250) — matrix operand order repaired and checked by reading, not compiled, not run on hardware
//
#include <hip/hip_runtime.h>
#include <math.h>

// ---------------------------------------------------------------------------
// SEAC (RandLA-style LocSE + attentive pooling + conv + skip), fused for
// MI455X / gfx1250. All four matmuls run on v_wmma_f32_16x16x32_f16.
// Conv GEMM: multi-accumulator (9 N-tiles/wave) to avoid WMMA RAW hazards and
// redundant LDS A-fragment loads. pc staged via global_load_async_to_lds.
// ---------------------------------------------------------------------------

typedef __attribute__((ext_vector_type(16))) _Float16 v16h;
typedef __attribute__((ext_vector_type(8)))  _Float16 v8h;
typedef __attribute__((ext_vector_type(4)))  _Float16 v4h;
typedef __attribute__((ext_vector_type(8)))  float    v8f;
typedef __attribute__((ext_vector_type(4)))  float    v4f;

#define KN        17          // 1+K neighbors
#define PDIMS     2
#define CHX       53          // rppe channels
#define DSX       32
#define DAX       64
#define DOX       32
#define CONV_IN   1090
#define CONV_INP  1120        // padded K for conv GEMM
#define CONV_OUT  544
#define CONV_NP   576         // padded N (36 tiles of 16) for even wave split
#define PTS       16          // points per block
#define ROWS      (PTS * KN)  // 272 (p,j) rows, exactly 17 M-tiles of 16

// f16 transposed-weight workspace layout (in halves). Layout is [n][k_pad]
// (k contiguous) so a WMMA B-fragment lane reads 16 consecutive halves.
#define OFF_MLP   0                       // [32][64]
#define OFF_ATT   2048                    // [64][64]
#define OFF_SKIP  6144                    // [32][64]
#define OFF_CONV  8192                    // [576][1120] (rows 544..575 zero)
#define WS_HALVES (8192 + CONV_NP * CONV_INP)    // 653312 halves ~ 1.25 MB

// dynamic LDS layout (bytes)
#define SM_A      0                       // 272*64*2 f16 rppe, then 272*64*4 f32 logits
#define SM_B      69632                   // 272*64*2 f16: se, later skip_in
#define SM_C      104448                  // 16*1120*2 f16: cat
#define SM_D      140288                  // 16*17*2 f32: pc stage (async target)
#define SM_TOTAL  142464

// ---- WMMA fragment loaders (16-bit, contiguous k order, same for A and B) ----

// A (16x32, MxK) from row-major f16 [m][k], stride ld halves.
// lane<16: halves0-15 = K 0..15; lane>=16: K 16..31.
__device__ __forceinline__ v16h load_a16(const _Float16* base, int m, int k0, int ld) {
  const int lane = threadIdx.x & 31;
  const int koff = (lane < 16) ? 0 : 16;
  const _Float16* p = base + m * ld + k0 + koff;
  return *(const v16h*)(p);
}

// B (32x16, KxN) from transposed weights [n][k], stride ld halves.
// lane L holds column n = L%16; K run = k0 + (L<16 ? 0..15 : 16..31).
__device__ __forceinline__ v16h load_b16(const _Float16* wt, int n0, int k0, int ld) {
  const int lane = threadIdx.x & 31;
  const _Float16* p = wt + (n0 + (lane & 15)) * ld + k0 + ((lane < 16) ? 0 : 16);
  return *(const v16h*)(p);
}

__device__ __forceinline__ v8f wmma16(v16h a, v16h b, v8f c) {
  return __builtin_amdgcn_wmma_f32_16x16x32_f16(false, a, false, b, (short)0, c,
                                                false, false);
}

// ---------------------------------------------------------------------------
// Weight prep: f32 -> f16, transpose to [n][k_pad], zero-pad K and N.
// ---------------------------------------------------------------------------
__global__ void seac_prep(const float* __restrict__ Wm, const float* __restrict__ Wa,
                          const float* __restrict__ Wc, const float* __restrict__ Wsk,
                          _Float16* __restrict__ wt) {
  int t = blockIdx.x * blockDim.x + threadIdx.x;
  if (t < 2048) {                                   // W_mlp (53x32) -> [32][64]
    int n = t >> 6, k = t & 63;
    wt[OFF_MLP + t] = (_Float16)((k < CHX) ? Wm[k * DSX + n] : 0.f);
  } else if (t < 6144) {                            // W_att (64x64) -> [64][64]
    int u = t - 2048, n = u >> 6, k = u & 63;
    wt[OFF_ATT + u] = (_Float16)Wa[k * DAX + n];
  } else if (t < 8192) {                            // W_skip (64x32) -> [32][64]
    int u = t - 6144, n = u >> 6, k = u & 63;
    wt[OFF_SKIP + u] = (_Float16)Wsk[k * DOX + n];
  } else if (t < WS_HALVES) {                       // W_conv (1090x544) -> [576][1120]
    int u = t - 8192, n = u / CONV_INP, k = u % CONV_INP;
    float v = (n < CONV_OUT && k < CONV_IN) ? Wc[k * CONV_OUT + n] : 0.f;
    wt[OFF_CONV + u] = (_Float16)v;
  }
}

// ---------------------------------------------------------------------------
// Fused kernel: 1 block (4 waves) per 16 points.
// ---------------------------------------------------------------------------
__global__ __launch_bounds__(128, 1) void seac_fused(
    const float* __restrict__ pc, const float* __restrict__ feats,
    const float* __restrict__ b_mlp, const float* __restrict__ b_att,
    const float* __restrict__ b_conv, const float* __restrict__ b_skip,
    const _Float16* __restrict__ wt, float* __restrict__ out) {
  extern __shared__ char smem[];
  _Float16* rppe   = (_Float16*)(smem + SM_A);   // [272][64] f16
  float*    logits = (float*)   (smem + SM_A);   // [272][64] f32 (reuses rppe)
  _Float16* seB    = (_Float16*)(smem + SM_B);   // [272][64] f16: se, then skip_in
  _Float16* catL   = (_Float16*)(smem + SM_C);   // [16][1120] f16
  float*    pcS    = (float*)   (smem + SM_D);   // [16][17][2] f32

  const int tid  = threadIdx.x;
  const int lane = tid & 31;
  const int wv   = tid >> 5;
  const int bn0  = blockIdx.x * PTS;             // first flat (b,n) point

  // 1. stage pc (raw f32, 2176 B) via async global->LDS copy (ASYNCcnt path)
  if (wv == 0) {
    const unsigned long long gbase =
        (unsigned long long)(uintptr_t)(pc + (long)bn0 * KN * PDIMS);
    const unsigned lbase = (unsigned)(uintptr_t)pcS;
#pragma unroll
    for (int k = 0; k < 5; ++k) {
      int idx = k * 32 + lane;                   // 136 x 16B chunks
      if (idx < (PTS * KN * PDIMS * 4) / 16) {
        unsigned           ldst = lbase + (unsigned)idx * 16u;
        unsigned long long gsrc = gbase + (unsigned long long)idx * 16ull;
        asm volatile("global_load_async_to_lds_b128 %0, %1, off"
                     :: "v"(ldst), "v"(gsrc) : "memory");
      }
    }
    asm volatile("s_wait_asynccnt 0" ::: "memory");
  }
  __syncthreads();

  // 2. rppe rows (f16, K padded to 64) + feats half of se (vectorized)
  for (int r = tid; r < ROWS; r += 128) {
    int p = r / KN, j = r % KN;
    float xj = pcS[p * 34 + j * 2 + 0];
    float yj = pcS[p * 34 + j * 2 + 1];
    float nj = sqrtf(xj * xj + yj * yj);
    _Float16* row = rppe + r * 64;
    row[0] = (_Float16)xj;
    row[1] = (_Float16)yj;
    for (int i = 0; i < KN; ++i) {
      float xi = pcS[p * 34 + i * 2 + 0];
      float yi = pcS[p * 34 + i * 2 + 1];
      row[2 + 3 * i] = (_Float16)(xi - xj);
      row[3 + 3 * i] = (_Float16)(yi - yj);
      row[4 + 3 * i] = (_Float16)nj;
    }
    for (int c = CHX; c < 64; ++c) row[c] = (_Float16)0.f;
  }
  {
    const v4f* fv = (const v4f*)(feats + (long)bn0 * KN * DSX);
    for (int i = tid; i < ROWS * 8; i += 128) {    // 8 float4 per row
      int r = i >> 3, q = i & 7;
      v4f v = fv[(long)r * 8 + q];
      v4h h;
#pragma unroll
      for (int e = 0; e < 4; ++e) h[e] = (_Float16)v[e];
      *(v4h*)(seB + r * 64 + 32 + q * 4) = h;
    }
  }
  __syncthreads();

  // 3. MLP GEMM: (272x64)x(64x32) -> relu -> se[:, 0:32]
  for (int t = wv; t < 34; t += 4) {
    int mt = t >> 1, nt = t & 1;
    v8f acc = {0.f, 0.f, 0.f, 0.f, 0.f, 0.f, 0.f, 0.f};
#pragma unroll
    for (int k0 = 0; k0 < 64; k0 += 32) {
      v16h a = load_a16(rppe, mt * 16 + (lane & 15), k0, 64);
      v16h b = load_b16(wt + OFF_MLP, nt * 16, k0, 64);
      acc = wmma16(a, b, acc);
    }
    int n = nt * 16 + (lane & 15);
    float bv = b_mlp[n];
#pragma unroll
    for (int r = 0; r < 8; ++r) {
      int m = mt * 16 + r + ((lane < 16) ? 0 : 8);
      seB[m * 64 + n] = (_Float16)fmaxf(acc[r] + bv, 0.f);
    }
  }
  __syncthreads();

  // 4. attention logits: (272x64)x(64x64) -> f32 LDS (reuses rppe region)
  for (int t = wv; t < 68; t += 4) {
    int mt = t >> 2, nt = t & 3;
    v8f acc = {0.f, 0.f, 0.f, 0.f, 0.f, 0.f, 0.f, 0.f};
#pragma unroll
    for (int k0 = 0; k0 < 64; k0 += 32) {
      v16h a = load_a16(seB, mt * 16 + (lane & 15), k0, 64);
      v16h b = load_b16(wt + OFF_ATT, nt * 16, k0, 64);
      acc = wmma16(a, b, acc);
    }
    int n = nt * 16 + (lane & 15);
    float bv = b_att[n];
#pragma unroll
    for (int r = 0; r < 8; ++r) {
      int m = mt * 16 + r + ((lane < 16) ? 0 : 8);
      logits[m * 64 + n] = acc[r] + bv;
    }
  }
  __syncthreads();

  // 5. per-row softmax over 64 channels, * se -> cat[p][2 + j*64 + c]
  for (int r = tid; r < ROWS; r += 128) {
    int p = r / KN, j = r % KN;
    const float* lr = logits + r * 64;
    float mx = -1e30f;
    for (int c = 0; c < 64; ++c) mx = fmaxf(mx, lr[c]);
    float s = 0.f;
    for (int c = 0; c < 64; ++c) s += __expf(lr[c] - mx);
    float inv = 1.f / s;
    _Float16* cr = catL + p * CONV_INP + 2 + j * 64;
    for (int c = 0; c < 64; ++c)
      cr[c] = (_Float16)(__expf(lr[c] - mx) * inv * (float)seB[r * 64 + c]);
  }
  for (int p = tid; p < PTS; p += 128) {
    catL[p * CONV_INP + 0] = (_Float16)pcS[p * 34 + 0];
    catL[p * CONV_INP + 1] = (_Float16)pcS[p * 34 + 1];
  }
  for (int i = tid; i < PTS * (CONV_INP - CONV_IN); i += 128) {
    int p = i / 30, c = i % 30;
    catL[p * CONV_INP + CONV_IN + c] = (_Float16)0.f;
  }
  __syncthreads();

  // 6a. feats part of skip_in (seB cols 0..31) -- disjoint from conv epilogue
  {
    const v4f* fv = (const v4f*)(feats + (long)bn0 * KN * DSX);
    for (int i = tid; i < ROWS * 8; i += 128) {
      int r = i >> 3, q = i & 7;
      v4f v = fv[(long)r * 8 + q];
      v4h h;
#pragma unroll
      for (int e = 0; e < 4; ++e) h[e] = (_Float16)v[e];
      *(v4h*)(seB + r * 64 + q * 4) = h;
    }
  }

  // 6b. conv GEMM: (16x1120)x(1120x576); 9 N-tile accumulators per wave.
  //     Epilogue writes relu(res) straight into skip_in (seB cols 32..63).
  {
    v8f acc[9];
#pragma unroll
    for (int i = 0; i < 9; ++i)
      acc[i] = (v8f){0.f, 0.f, 0.f, 0.f, 0.f, 0.f, 0.f, 0.f};
    for (int k0 = 0; k0 < CONV_INP; k0 += 32) {
      __builtin_prefetch(
          (const void*)(wt + OFF_CONV + (wv * 16 + (lane & 15)) * CONV_INP + k0 + 64),
          0, 0);
      v16h a = load_a16(catL, (lane & 15), k0, CONV_INP);
#pragma unroll
      for (int i = 0; i < 9; ++i) {
        int nt = wv + 4 * i;                        // disjoint across 4 waves, 0..35
        v16h b = load_b16(wt + OFF_CONV, nt * 16, k0, CONV_INP);
        acc[i] = wmma16(a, b, acc[i]);
      }
    }
#pragma unroll
    for (int i = 0; i < 9; ++i) {
      int nt = wv + 4 * i;
      if (nt < 34) {                                // wave-uniform guard (pad tiles)
        int col = nt * 16 + (lane & 15);            // 0..543 = j*32 + c2
        float bv = b_conv[col];
        int j = col >> 5, c2 = col & 31;
#pragma unroll
        for (int r = 0; r < 8; ++r) {
          int p = r + ((lane < 16) ? 0 : 8);        // point index (M rows = points)
          seB[(p * KN + j) * 64 + 32 + c2] = (_Float16)fmaxf(acc[i][r] + bv, 0.f);
        }
      }
    }
  }
  __syncthreads();

  // 7. skip GEMM: (272x64)x(64x32) -> relu -> out (f32)
  for (int t = wv; t < 34; t += 4) {
    int mt = t >> 1, nt = t & 1;
    v8f acc = {0.f, 0.f, 0.f, 0.f, 0.f, 0.f, 0.f, 0.f};
#pragma unroll
    for (int k0 = 0; k0 < 64; k0 += 32) {
      v16h a = load_a16(seB, mt * 16 + (lane & 15), k0, 64);
      v16h b = load_b16(wt + OFF_SKIP, nt * 16, k0, 64);
      acc = wmma16(a, b, acc);
    }
    int n = nt * 16 + (lane & 15);
    float bv = b_skip[n];
#pragma unroll
    for (int r = 0; r < 8; ++r) {
      int m = mt * 16 + r + ((lane < 16) ? 0 : 8);
      out[((long)bn0 * KN + m) * 32 + n] = fmaxf(acc[r] + bv, 0.f);
    }
  }
}

// ---------------------------------------------------------------------------
extern "C" void kernel_launch(void* const* d_in, const int* in_sizes, int n_in,
                              void* d_out, int out_size, void* d_ws, size_t ws_size,
                              hipStream_t stream) {
  const float* pc    = (const float*)d_in[0];
  const float* feats = (const float*)d_in[1];
  const float* Wm    = (const float*)d_in[2];
  const float* bm    = (const float*)d_in[3];
  const float* Wa    = (const float*)d_in[4];
  const float* ba    = (const float*)d_in[5];
  const float* Wc    = (const float*)d_in[6];
  const float* bc    = (const float*)d_in[7];
  const float* Wsk   = (const float*)d_in[8];
  const float* bsk   = (const float*)d_in[9];
  _Float16*    wt    = (_Float16*)d_ws;

  const int BN = in_sizes[0] / (KN * PDIMS);       // 65536 flat points

  const int prep_threads = 256;
  const int prep_blocks  = (WS_HALVES + prep_threads - 1) / prep_threads;
  seac_prep<<<prep_blocks, prep_threads, 0, stream>>>(Wm, Wa, Wc, Wsk, wt);

  (void)hipFuncSetAttribute(reinterpret_cast<const void*>(seac_fused),
                            hipFuncAttributeMaxDynamicSharedMemorySize, SM_TOTAL);
  seac_fused<<<BN / PTS, 128, SM_TOTAL, stream>>>(pc, feats, bm, ba, bc, bsk, wt,
                                                  (float*)d_out);
}
